// non_dir_model_3066606649907
// MI455X (gfx1250) — matrix operand order repaired and checked by reading, not compiled, not run on hardware
//
#include <hip/hip_runtime.h>
#include <hip/hip_bf16.h>

// Model dims (fixed by the reference)
#define B_    8
#define T_    32
#define C_    1024
#define H_    128
#define D_    32
#define ROWS_ (C_ * B_)   // 8192

typedef __attribute__((ext_vector_type(16))) __bf16 v16bf;
typedef __attribute__((ext_vector_type(8)))  float  v8f;

union Frag { v16bf v; uint4 q[2]; };

__device__ __forceinline__ unsigned short f2bf(float f) {
  union { float f; unsigned u; } x; x.f = f;
  unsigned r = x.u + 0x7FFFu + ((x.u >> 16) & 1u);   // round-to-nearest-even
  return (unsigned short)(r >> 16);
}
__device__ __forceinline__ float bf2f(unsigned short h) {
  union { unsigned u; float f; } x; x.u = ((unsigned)h) << 16;
  return x.f;
}

// ---------------------------------------------------------------------------
// Generic bf16 WMMA GEMM (small-N weight matmuls):
//   C(MxN) = A(MxK, row-major bf16) * Bt^T  (+bias, relu), f32 and/or bf16 out
// block = 128 threads = 4 waves; wave tile 32x32 (2x2 WMMA frags).
// Requires M % 128 == 0, N % 32 == 0, K % 32 == 0 -> EXEC all-ones at WMMA.
// ---------------------------------------------------------------------------
__global__ __launch_bounds__(128)
void gemm_bf16(const unsigned short* __restrict__ A,
               const unsigned short* __restrict__ Bt,
               int M, int N, int K, int lda, int ldb,
               const float* __restrict__ bias, int relu,
               float* __restrict__ Cf, unsigned short* __restrict__ Cb, int ldc)
{
  const int lane = threadIdx.x & 31;
  const int wave = threadIdx.x >> 5;
  const int hh   = lane >> 4;     // lane half (0/1)
  const int l16  = lane & 15;
  const int m0 = blockIdx.y * 128 + wave * 32;
  const int n0 = blockIdx.x * 32;

  v8f acc00 = {}; v8f acc01 = {}; v8f acc10 = {}; v8f acc11 = {};

  const unsigned short* Arow0 = A  + (size_t)(m0 +      l16) * lda + hh * 8;
  const unsigned short* Arow1 = A  + (size_t)(m0 + 16 + l16) * lda + hh * 8;
  const unsigned short* Brow0 = Bt + (size_t)(n0 +      l16) * ldb + hh * 8;
  const unsigned short* Brow1 = Bt + (size_t)(n0 + 16 + l16) * ldb + hh * 8;

  for (int k0 = 0; k0 < K; k0 += 32) {
    Frag a0, a1, b0, b1;
    a0.q[0] = *reinterpret_cast<const uint4*>(Arow0 + k0);
    a0.q[1] = *reinterpret_cast<const uint4*>(Arow0 + k0 + 16);
    a1.q[0] = *reinterpret_cast<const uint4*>(Arow1 + k0);
    a1.q[1] = *reinterpret_cast<const uint4*>(Arow1 + k0 + 16);
    b0.q[0] = *reinterpret_cast<const uint4*>(Brow0 + k0);
    b0.q[1] = *reinterpret_cast<const uint4*>(Brow0 + k0 + 16);
    b1.q[0] = *reinterpret_cast<const uint4*>(Brow1 + k0);
    b1.q[1] = *reinterpret_cast<const uint4*>(Brow1 + k0 + 16);

    if (k0 + 32 < K) {
      __builtin_prefetch(Arow0 + k0 + 32, 0, 1);
      __builtin_prefetch(Brow0 + k0 + 32, 0, 1);
    }

    acc00 = __builtin_amdgcn_wmma_f32_16x16x32_bf16(false, a0.v, false, b0.v, (short)0, acc00, false, false);
    acc01 = __builtin_amdgcn_wmma_f32_16x16x32_bf16(false, a0.v, false, b1.v, (short)0, acc01, false, false);
    acc10 = __builtin_amdgcn_wmma_f32_16x16x32_bf16(false, a1.v, false, b0.v, (short)0, acc10, false, false);
    acc11 = __builtin_amdgcn_wmma_f32_16x16x32_bf16(false, a1.v, false, b1.v, (short)0, acc11, false, false);
  }

  v8f accs[2][2] = { { acc00, acc01 }, { acc10, acc11 } };
  for (int mi = 0; mi < 2; ++mi) {
    for (int ni = 0; ni < 2; ++ni) {
      const int n  = n0 + ni * 16 + l16;
      const float bv = bias ? bias[n] : 0.0f;
      const int mb = m0 + mi * 16 + hh * 8;
      for (int v = 0; v < 8; ++v) {
        float val = accs[mi][ni][v] + bv;
        if (relu && val < 0.0f) val = 0.0f;
        const size_t off = (size_t)(mb + v) * ldc + n;
        if (Cf) Cf[off] = val;
        if (Cb) Cb[off] = f2bf(val);
      }
    }
  }
}

// ---------------------------------------------------------------------------
// Big-tile bf16 WMMA GEMM for the Laplacian aggregations (~73% of FLOPs):
// block = 128 threads = 2x2 waves; wave tile 64x64 (4x4 frags, 16 WMMA/slab,
// 16 b128 loads/slab -> 1 VMEM per WMMA, 64 flop/B from L2).
// No bias/relu; bf16 output. M % 128 == 0, N % 128 == 0, K % 32 == 0.
// ---------------------------------------------------------------------------
__global__ __launch_bounds__(128)
void gemm_bf16_big(const unsigned short* __restrict__ A,
                   const unsigned short* __restrict__ Bt,
                   int K, int lda, int ldb,
                   unsigned short* __restrict__ Cb, int ldc)
{
  const int lane = threadIdx.x & 31;
  const int wave = threadIdx.x >> 5;   // 0..3 -> 2x2 wave grid
  const int wm = wave >> 1, wn = wave & 1;
  const int hh = lane >> 4, l16 = lane & 15;
  const int m0 = blockIdx.y * 128 + wm * 64;
  const int n0 = blockIdx.x * 128 + wn * 64;

  v8f acc[4][4] = {};
  const unsigned short* Ar[4];
  const unsigned short* Br[4];
#pragma unroll
  for (int i = 0; i < 4; ++i) {
    Ar[i] = A  + (size_t)(m0 + i * 16 + l16) * lda + hh * 8;
    Br[i] = Bt + (size_t)(n0 + i * 16 + l16) * ldb + hh * 8;
  }

  for (int k0 = 0; k0 < K; k0 += 32) {
    Frag a[4], b[4];
#pragma unroll
    for (int i = 0; i < 4; ++i) {
      a[i].q[0] = *reinterpret_cast<const uint4*>(Ar[i] + k0);
      a[i].q[1] = *reinterpret_cast<const uint4*>(Ar[i] + k0 + 16);
      b[i].q[0] = *reinterpret_cast<const uint4*>(Br[i] + k0);
      b[i].q[1] = *reinterpret_cast<const uint4*>(Br[i] + k0 + 16);
    }
    if (k0 + 32 < K) {
      __builtin_prefetch(Ar[0] + k0 + 32, 0, 1);
      __builtin_prefetch(Br[0] + k0 + 32, 0, 1);
    }
#pragma unroll
    for (int mi = 0; mi < 4; ++mi)
#pragma unroll
      for (int ni = 0; ni < 4; ++ni)
        acc[mi][ni] = __builtin_amdgcn_wmma_f32_16x16x32_bf16(
            false, a[mi].v, false, b[ni].v, (short)0, acc[mi][ni], false, false);
  }

#pragma unroll
  for (int mi = 0; mi < 4; ++mi)
#pragma unroll
    for (int ni = 0; ni < 4; ++ni) {
      const int n  = n0 + ni * 16 + l16;
      const int mb = m0 + mi * 16 + hh * 8;
#pragma unroll
      for (int v = 0; v < 8; ++v)
        Cb[(size_t)(mb + v) * ldc + n] = f2bf(acc[mi][ni][v]);
    }
}

// ---------------------------------------------------------------------------
// dinv_f[t,i] = 1/sqrt(1 + sum_j adj[t,i,j]) ; dinv_b: column sums.
// ---------------------------------------------------------------------------
__global__ void laplace_dinv(const float* __restrict__ adj,
                             float* __restrict__ dinvf, float* __restrict__ dinvb)
{
  int idx = blockIdx.x * blockDim.x + threadIdx.x;
  if (idx >= T_ * C_) return;
  int t = idx >> 10, i = idx & (C_ - 1);
  const float* At = adj + ((size_t)t << 20);
  float sf = 1.0f, sb = 1.0f;
  for (int j = 0; j < C_; ++j) {
    sf += At[(size_t)i * C_ + j];
    sb += At[(size_t)j * C_ + i];
  }
  dinvf[idx] = rsqrtf(sf);
  dinvb[idx] = rsqrtf(sb);
}

// Lf[i,j] = (adj[t,i,j]+I) * df[i]*df[j];  Lb[i,j] = (adj[t,j,i]+I) * db[i]*db[j]
__global__ void build_L(const float* __restrict__ adj,
                        const float* __restrict__ dinvf, const float* __restrict__ dinvb,
                        unsigned short* __restrict__ Lf, unsigned short* __restrict__ Lb, int t)
{
  int idx = blockIdx.x * blockDim.x + threadIdx.x;
  if (idx >= C_ * C_) return;
  int i = idx >> 10, j = idx & (C_ - 1);
  const float* At = adj + ((size_t)t << 20);
  const float* df = dinvf + t * C_;
  const float* db = dinvb + t * C_;
  float eye = (i == j) ? 1.0f : 0.0f;
  Lf[idx] = f2bf((At[(size_t)i * C_ + j] + eye) * df[i] * df[j]);
  Lb[idx] = f2bf((At[(size_t)j * C_ + i] + eye) * db[i] * db[j]);
}

// XT[(b*64+f), c] = bf16(x[b,t,c,f]) with x = concat(in[...,:32], in[...,128:160])
__global__ void build_XT(const float* __restrict__ inp, unsigned short* __restrict__ XT, int t)
{
  int idx = blockIdx.x * blockDim.x + threadIdx.x;
  if (idx >= 512 * C_) return;
  int row = idx >> 10, c = idx & (C_ - 1);
  int b = row >> 6, f = row & 63;
  int fs = (f < 32) ? f : (96 + f);  // 128 + (f-32)
  XT[idx] = f2bf(inp[((((size_t)b * T_ + t) * C_ + c) * 160) + fs]);
}

// Wt[n*K + k] = bf16(W[k*N + n])
__global__ void wt_transpose(const float* __restrict__ W, unsigned short* __restrict__ Wt,
                             int K, int N)
{
  int idx = blockIdx.x * blockDim.x + threadIdx.x;
  if (idx >= K * N) return;
  int n = idx / K, k = idx - n * K;
  Wt[idx] = f2bf(W[(size_t)k * N + n]);
}

// H (flat (C,B,H) = (C, 1024)) -> HT[(b*128+f), c]
__global__ void h_transpose(const unsigned short* __restrict__ Hm, unsigned short* __restrict__ HT)
{
  int idx = blockIdx.x * blockDim.x + threadIdx.x;
  if (idx >= C_ * 1024) return;
  int c = idx >> 10, bf = idx & 1023;
  HT[(size_t)bf * C_ + c] = Hm[idx];
}

// GRU pointwise: xi=tmp (GI), "h"=hs (GH, HS). Writes new hidden H + transpose.
__global__ void gru_update(const float* __restrict__ GI, const float* __restrict__ GH,
                           const unsigned short* __restrict__ HSb,
                           unsigned short* __restrict__ Hm, unsigned short* __restrict__ HT)
{
  int idx = blockIdx.x * blockDim.x + threadIdx.x;
  if (idx >= ROWS_ * H_) return;
  int r = idx >> 7, f = idx & (H_ - 1);
  int c = r >> 3, b = r & 7;
  size_t g = (size_t)r * 384 + f;
  float ir = GI[g], iz = GI[g + 128], in = GI[g + 256];
  float hr = GH[g], hz = GH[g + 128], hn = GH[g + 256];
  float hs = bf2f(HSb[idx]);
  float rg = 1.0f / (1.0f + __expf(-(ir + hr)));
  float zg = 1.0f / (1.0f + __expf(-(iz + hz)));
  float ng = tanhf(in + rg * hn);
  float hnew = (1.0f - zg) * ng + zg * hs;
  unsigned short hb = f2bf(hnew);
  Hm[idx] = hb;
  HT[(size_t)(b * H_ + f) * C_ + c] = hb;
}

__global__ void build_mask(const int* __restrict__ cells, int ncells, int* __restrict__ mask)
{
  int c = threadIdx.x;          // single block of 1024
  mask[c] = 0;
  __syncthreads();
  if (c < ncells) mask[cells[c]] = 1;
}

// out[b, s=t-4, c, d] = mask[c] ? x[b, t+1, c, d] : PRED[(c*B+b), d]
__global__ void write_output(const float* __restrict__ PRED, const float* __restrict__ inp,
                             const int* __restrict__ mask, float* __restrict__ out, int t)
{
  int idx = blockIdx.x * blockDim.x + threadIdx.x;
  if (idx >= B_ * C_ * D_) return;
  int b = idx >> 15, rem = idx & 32767;
  int c = rem >> 5, d = rem & 31;
  int s = t - 4;
  float val;
  if (mask[c]) val = inp[(((size_t)b * T_ + (t + 1)) * C_ + c) * 160 + d];
  else         val = PRED[((size_t)c * B_ + b) * D_ + d];
  out[(((size_t)b * 27 + s) * C_ + c) * D_ + d] = val;
}

// ---------------------------------------------------------------------------
extern "C" void kernel_launch(void* const* d_in, const int* in_sizes, int n_in,
                              void* d_out, int out_size, void* d_ws, size_t ws_size,
                              hipStream_t stream)
{
  (void)n_in; (void)out_size; (void)ws_size;

  const float* in_data  = (const float*)d_in[0];
  const float* adj      = (const float*)d_in[1];
  const float* init_W   = (const float*)d_in[2];
  const float* init_b   = (const float*)d_in[3];
  const float* fe_fwd_W = (const float*)d_in[4];
  const float* fe_fwd_b = (const float*)d_in[5];
  const float* fe_bwd_W = (const float*)d_in[6];
  const float* fe_bwd_b = (const float*)d_in[7];
  const float* fwd_W    = (const float*)d_in[8];
  const float* fwd_b    = (const float*)d_in[9];
  const float* bwd_W    = (const float*)d_in[10];
  const float* bwd_b    = (const float*)d_in[11];
  const float* merge_W  = (const float*)d_in[12];
  const float* merge_b  = (const float*)d_in[13];
  const float* gru_Wih  = (const float*)d_in[14];
  const float* gru_bih  = (const float*)d_in[15];
  const float* gru_Whh  = (const float*)d_in[16];
  const float* gru_bhh  = (const float*)d_in[17];
  const float* out_W    = (const float*)d_in[18];
  const float* out_b    = (const float*)d_in[19];
  const int*   cells    = (const int*)d_in[20];
  float* out = (float*)d_out;

  // ---- carve workspace (all 256B aligned) -------------------------------
  char* p = (char*)d_ws;
  auto take = [&](size_t bytes) -> char* {
    char* r = p; p += (bytes + 255) & ~(size_t)255; return r;
  };
  float*          d_dinvf  = (float*)take(T_ * C_ * 4);
  float*          d_dinvb  = (float*)take(T_ * C_ * 4);
  int*            d_mask   = (int*)take(C_ * 4);
  unsigned short* d_Lf     = (unsigned short*)take((size_t)C_ * C_ * 2);
  unsigned short* d_Lb     = (unsigned short*)take((size_t)C_ * C_ * 2);
  unsigned short* d_XT     = (unsigned short*)take((size_t)512 * C_ * 2);
  unsigned short* d_Aggf   = (unsigned short*)take((size_t)C_ * 512 * 2);
  unsigned short* d_Aggb   = (unsigned short*)take((size_t)C_ * 512 * 2);
  unsigned short* d_TMP    = (unsigned short*)take((size_t)ROWS_ * 64 * 2);
  unsigned short* d_AggHf  = (unsigned short*)take((size_t)ROWS_ * H_ * 2);
  unsigned short* d_AggHb  = (unsigned short*)take((size_t)ROWS_ * H_ * 2);
  unsigned short* d_HScat  = (unsigned short*)take((size_t)ROWS_ * 256 * 2);
  unsigned short* d_HS     = (unsigned short*)take((size_t)ROWS_ * H_ * 2);
  float*          d_GI     = (float*)take((size_t)ROWS_ * 384 * 4);
  float*          d_GH     = (float*)take((size_t)ROWS_ * 384 * 4);
  unsigned short* d_H      = (unsigned short*)take((size_t)ROWS_ * H_ * 2);
  unsigned short* d_HT     = (unsigned short*)take((size_t)C_ * 1024 * 2);
  float*          d_PRED   = (float*)take((size_t)ROWS_ * D_ * 4);
  unsigned short* d_initWt = (unsigned short*)take(64 * 128 * 2);
  unsigned short* d_feFwdWt= (unsigned short*)take(64 * 32 * 2);
  unsigned short* d_feBwdWt= (unsigned short*)take(64 * 32 * 2);
  unsigned short* d_fwdWt  = (unsigned short*)take(128 * 128 * 2);
  unsigned short* d_bwdWt  = (unsigned short*)take(128 * 128 * 2);
  unsigned short* d_mergeWt= (unsigned short*)take(256 * 128 * 2);
  unsigned short* d_wihT   = (unsigned short*)take(64 * 384 * 2);
  unsigned short* d_whhT   = (unsigned short*)take(128 * 384 * 2);
  unsigned short* d_outWt  = (unsigned short*)take(128 * 32 * 2);

  // ---- setup ------------------------------------------------------------
  auto wtT = [&](const float* W, unsigned short* Wt, int K, int N) {
    int n = K * N;
    wt_transpose<<<(n + 255) / 256, 256, 0, stream>>>(W, Wt, K, N);
  };
  wtT(init_W,   d_initWt,  64, 128);
  wtT(fe_fwd_W, d_feFwdWt, 64, 32);
  wtT(fe_bwd_W, d_feBwdWt, 64, 32);
  wtT(fwd_W,    d_fwdWt,   128, 128);
  wtT(bwd_W,    d_bwdWt,   128, 128);
  wtT(merge_W,  d_mergeWt, 256, 128);
  wtT(gru_Wih,  d_wihT,    64, 384);
  wtT(gru_Whh,  d_whhT,    128, 384);
  wtT(out_W,    d_outWt,   128, 32);

  laplace_dinv<<<(T_ * C_ + 255) / 256, 256, 0, stream>>>(adj, d_dinvf, d_dinvb);
  build_mask<<<1, 1024, 0, stream>>>(cells, in_sizes[20], d_mask);

  auto gemm = [&](const unsigned short* A, const unsigned short* Bt, int M, int N, int K,
                  int lda, int ldb, const float* bias, int relu,
                  float* Cf, unsigned short* Cb, int ldc) {
    dim3 g(N / 32, M / 128);
    gemm_bf16<<<g, 128, 0, stream>>>(A, Bt, M, N, K, lda, ldb, bias, relu, Cf, Cb, ldc);
  };
  // Aggregation GEMMs: M = C_, K = C_, N in {512, 1024}; 64x64 wave tiles.
  auto gemm_agg = [&](const unsigned short* L, const unsigned short* Bt, int N,
                      unsigned short* Cb) {
    dim3 g(N / 128, C_ / 128);
    gemm_bf16_big<<<g, 128, 0, stream>>>(L, Bt, C_, C_, C_, Cb, N);
  };

  // ---- hidden0 = relu((Lf[0] @ X0) @ init_W + init_b) -------------------
  build_L <<<4096, 256, 0, stream>>>(adj, d_dinvf, d_dinvb, d_Lf, d_Lb, 0);
  build_XT<<<2048, 256, 0, stream>>>(in_data, d_XT, 0);
  gemm_agg(d_Lf, d_XT, 512, d_Aggf);
  gemm(d_Aggf, d_initWt, ROWS_, 128, 64, 64, 64, init_b, 1, nullptr, d_H, 128);
  h_transpose<<<4096, 256, 0, stream>>>(d_H, d_HT);

  // ---- scan: t = 0 .. 30 ------------------------------------------------
  for (int t = 0; t < T_ - 1; ++t) {
    build_L <<<4096, 256, 0, stream>>>(adj, d_dinvf, d_dinvb, d_Lf, d_Lb, t);
    build_XT<<<2048, 256, 0, stream>>>(in_data, d_XT, t);

    // feature GCNs: tmp = [relu(Lf@X @ feW_f + b) , relu(Lb@X @ feW_b + b)]
    gemm_agg(d_Lf, d_XT, 512, d_Aggf);
    gemm_agg(d_Lb, d_XT, 512, d_Aggb);
    gemm(d_Aggf, d_feFwdWt, ROWS_, 32, 64, 64, 64, fe_fwd_b, 1, nullptr, d_TMP,      64);
    gemm(d_Aggb, d_feBwdWt, ROWS_, 32, 64, 64, 64, fe_bwd_b, 1, nullptr, d_TMP + 32, 64);

    // hidden GCNs: hs = [relu(Lf@H @ fwdW) , relu(Lb@H @ bwdW)] @ mergeW + b
    gemm_agg(d_Lf, d_HT, 1024, d_AggHf);
    gemm_agg(d_Lb, d_HT, 1024, d_AggHb);
    gemm(d_AggHf, d_fwdWt,   ROWS_, 128, 128, 128, 128, fwd_b,   1, nullptr, d_HScat,       256);
    gemm(d_AggHb, d_bwdWt,   ROWS_, 128, 128, 128, 128, bwd_b,   1, nullptr, d_HScat + 128, 256);
    gemm(d_HScat, d_mergeWt, ROWS_, 128, 256, 256, 256, merge_b, 0, nullptr, d_HS,          128);

    // GRU gates
    gemm(d_TMP, d_wihT, ROWS_, 384, 64,  64,  64,  gru_bih, 0, d_GI, nullptr, 384);
    gemm(d_HS,  d_whhT, ROWS_, 384, 128, 128, 128, gru_bhh, 0, d_GH, nullptr, 384);
    gru_update<<<4096, 256, 0, stream>>>(d_GI, d_GH, d_HS, d_H, d_HT);

    // prediction + masked output for t >= INIT_LENGTH
    if (t >= 4) {
      gemm(d_H, d_outWt, ROWS_, 32, 128, 128, 128, out_b, 0, d_PRED, nullptr, 32);
      write_output<<<(B_ * C_ * D_ + 255) / 256, 256, 0, stream>>>(d_PRED, in_data, d_mask, out, t);
    }
  }
}
